// ParityGameGATConv_27075473834772
// MI455X (gfx1250) — matrix operand order repaired and checked by reading, not compiled, not run on hardware
//
#include <hip/hip_runtime.h>
#include <hip/hip_bf16.h>

typedef __attribute__((ext_vector_type(16))) _Float16 v16h;
typedef __attribute__((ext_vector_type(4)))  _Float16 v4h;
typedef __attribute__((ext_vector_type(8)))  float    v8f;

#define HDIM 128

// ---------------- small helpers ----------------
__device__ __forceinline__ float sigmoidf_(float x) { return 1.0f / (1.0f + expf(-x)); }

__device__ __forceinline__ unsigned f2ord(float f) {
  unsigned u = __float_as_uint(f);
  return (u & 0x80000000u) ? ~u : (u | 0x80000000u);
}
__device__ __forceinline__ float ord2f(unsigned u) {
  return (u & 0x80000000u) ? __uint_as_float(u & 0x7fffffffu) : __uint_as_float(~u);
}

// A-fragment K index for v_wmma_f32_16x16x32_f16 (contiguous order: k = 16*hi + e, matches B)
__device__ __forceinline__ int kA_idx(int hi, int e) {
  return hi * 16 + e;
}

// pack float4 -> 4 halves (optionally scaled by mask) and store to LDS
__device__ __forceinline__ void st4h(_Float16* dst, float4 v, float m) {
  v4h h;
  h[0] = (_Float16)(v.x * m);
  h[1] = (_Float16)(v.y * m);
  h[2] = (_Float16)(v.z * m);
  h[3] = (_Float16)(v.w * m);
  *(v4h*)dst = h;
}

// ---------------- fill kernels ----------------
__global__ void fill_f32(float* p, float v, long n) {
  long i = (long)blockIdx.x * blockDim.x + threadIdx.x;
  if (i < n) p[i] = v;
}
__global__ void fill_u32(unsigned* p, unsigned v, long n) {
  long i = (long)blockIdx.x * blockDim.x + threadIdx.x;
  if (i < n) p[i] = v;
}
__global__ void fill_from(float* p, const float* v, long n) {
  long i = (long)blockIdx.x * blockDim.x + threadIdx.x;
  if (i < n) p[i] = v[0];
}

// v[k] = sum_j we1[j,k] * ae1[j]   (we1: [128,8])
__global__ void compute_vsmall(const float* __restrict__ we1, const float* __restrict__ ae1,
                               float* __restrict__ v) {
  int k = threadIdx.x;
  if (k < 8) {
    float s = 0.0f;
    for (int j = 0; j < HDIM; ++j) s += we1[j * 8 + k] * ae1[j];
    v[k] = s;
  }
}

// encode: h[n,o] = sum_{k<16} x[n,k] * w1[o,k]
__global__ void encode_gemm(const float* __restrict__ x, const float* __restrict__ w,
                            float* __restrict__ h, int N) {
  long gid = (long)blockIdx.x * blockDim.x + threadIdx.x;
  if (gid >= (long)N * HDIM) return;
  int n = (int)(gid >> 7), o = (int)(gid & 127);
  float s = 0.0f;
  for (int k = 0; k < 16; ++k) s += x[(long)n * 16 + k] * w[(long)o * 16 + k];
  h[gid] = s;
}

// per-node attention dot products: sd[n] = h[n].a_s ; sd[N+n] = h[n].a_d
__global__ void node_dots(const float* __restrict__ h, const float* __restrict__ a_s,
                          const float* __restrict__ a_d, float* __restrict__ sd, int N) {
  int n = blockIdx.x * blockDim.x + threadIdx.x;
  if (n >= N) return;
  float s = 0.0f, d = 0.0f;
  for (int i = 0; i < HDIM; ++i) {
    float v = h[(long)n * HDIM + i];
    s += v * a_s[i];
    d += v * a_d[i];
  }
  sd[n] = s;
  sd[N + n] = d;
}

// e = leaky(sd_s[src] + sd_d[dst] (+ edge_attr.vsm)); atomicMax ordered into mord[dst]
__global__ void edge_score(float* __restrict__ ebuf, const float* __restrict__ sd,
                           const int* __restrict__ src, const int* __restrict__ dst,
                           const float* __restrict__ ea, const float* __restrict__ vsm,
                           unsigned* __restrict__ mord, int N, int E) {
  int e = blockIdx.x * blockDim.x + threadIdx.x;
  if (e >= E) return;
  float v = sd[src[e]] + sd[N + dst[e]];
  if (ea != nullptr) {
    float t = 0.0f;
    for (int k = 0; k < 8; ++k) t += ea[(long)e * 8 + k] * vsm[k];
    v += t;
  }
  v = (v > 0.0f) ? v : 0.2f * v;
  ebuf[e] = v;
  atomicMax(&mord[dst[e]], f2ord(v));
}

// ex = exp(e - m[dst]); denom[dst] += ex
__global__ void edge_exp(float* __restrict__ ebuf, const int* __restrict__ dst,
                         const unsigned* __restrict__ mord, float* __restrict__ denom, int E) {
  int e = blockIdx.x * blockDim.x + threadIdx.x;
  if (e >= E) return;
  float m = ord2f(mord[dst[e]]);
  float ex = expf(ebuf[e] - m);
  ebuf[e] = ex;
  atomicAdd(&denom[dst[e]], ex);
}

// out[dst,f] += h[src,f] * (ex / (denom[dst]+1e-16))
__global__ void edge_agg(float* __restrict__ out, const float* __restrict__ h,
                         const int* __restrict__ src, const int* __restrict__ dst,
                         const float* __restrict__ ebuf, const float* __restrict__ denom, int E) {
  long gid = (long)blockIdx.x * blockDim.x + threadIdx.x;
  if (gid >= (long)E * HDIM) return;
  int e = (int)(gid >> 7), f = (int)(gid & 127);
  int d = dst[e], s = src[e];
  float alpha = ebuf[e] / (denom[d] + 1e-16f);
  atomicAdd(&out[(long)d * HDIM + f], h[(long)s * HDIM + f] * alpha);
}

__global__ void bias_act(float* __restrict__ p, const float* __restrict__ b, long total,
                         int do_relu) {
  long gid = (long)blockIdx.x * blockDim.x + threadIdx.x;
  if (gid >= total) return;
  int f = (int)(gid & 127);
  float v = p[gid] + b[f];
  if (do_relu) v = fmaxf(v, 0.0f);
  p[gid] = v;
}

// LSTM gate nonlinearity (torch order i,f,g,o); c updated in place, h written
__global__ void lstm_gates(const float* __restrict__ g, float* __restrict__ c,
                           float* __restrict__ h, int N) {
  long gid = (long)blockIdx.x * blockDim.x + threadIdx.x;
  if (gid >= (long)N * HDIM) return;
  int n = (int)(gid >> 7), f = (int)(gid & 127);
  const float* gr = g + (long)n * 512;
  float ig = sigmoidf_(gr[f]);
  float fg = sigmoidf_(gr[128 + f]);
  float gg = tanhf(gr[256 + f]);
  float og = sigmoidf_(gr[384 + f]);
  float cv = fg * c[gid] + ig * gg;
  c[gid] = cv;
  h[gid] = og * tanhf(cv);
}

// score[n] += h[n,:] . jw[0:128]
__global__ void score_acc(float* __restrict__ score, const float* __restrict__ h,
                          const float* __restrict__ jw, int N) {
  int n = blockIdx.x * blockDim.x + threadIdx.x;
  if (n >= N) return;
  float s = 0.0f;
  for (int i = 0; i < HDIM; ++i) s += h[(long)n * HDIM + i] * jw[i];
  score[n] += s;
}

// softmax over L=2 layer scores, mix xs0/xs1
__global__ void zmix_kernel(const float* __restrict__ score, const float* __restrict__ xs0,
                            const float* __restrict__ xs1, float* __restrict__ out, int N) {
  long gid = (long)blockIdx.x * blockDim.x + threadIdx.x;
  if (gid >= (long)N * HDIM) return;
  int n = (int)(gid >> 7);
  float a0 = score[n], a1 = score[N + n];
  float m = fmaxf(a0, a1);
  float e0 = expf(a0 - m), e1 = expf(a1 - m);
  float inv = 1.0f / (e0 + e1);
  out[gid] = (e0 * xs0[gid] + e1 * xs1[gid]) * inv;
}

// ---------------- WMMA GEMM: C[M,ldc(cols)] (+=/=) act(A[M,K] @ W[Nout,K]^T + b0 + b1)
// block = 128 threads (4 waves); block tile = 64 rows x 128 cols; grid.y picks 128-col chunk.
__global__ __launch_bounds__(128) void gemm128_wmma(
    const float* __restrict__ A, const float* __restrict__ W,
    const float* __restrict__ bias0, const float* __restrict__ bias1,
    float* __restrict__ C, int M, int K, int ldc, int accumulate, int do_relu) {
  __shared__ _Float16 As[64][40];
  __shared__ _Float16 Bs[128][40];
  const int tid = threadIdx.x;
  const int lane = tid & 31;
  const int wave = tid >> 5;
  const int hi = lane >> 4;
  const int nlane = lane & 15;
  const int rowbase = blockIdx.x * 64;
  const int colbase = blockIdx.y * 128;
  const int Mc = M - 1;

  v8f acc[8];
  for (int i = 0; i < 8; ++i)
    for (int j = 0; j < 8; ++j) acc[i][j] = 0.0f;

  const int ktiles = K >> 5;
  for (int kt = 0; kt < ktiles; ++kt) {
    // stage A tile (64 rows x 32 k) as f16 via float4 loads (clamped rows; OOB harmless)
    for (int i = 0; i < 4; ++i) {
      int e = tid + i * 128;
      int r = e >> 3, k4 = e & 7;
      int row = rowbase + r;
      int rc = (row < M) ? row : Mc;
      float4 v = *(const float4*)(A + (long)rc * K + kt * 32 + k4 * 4);
      st4h(&As[r][k4 * 4], v, 1.0f);
    }
    // stage B tile (128 cols x 32 k) as f16 (W row-major [Nout,K])
    for (int i = 0; i < 8; ++i) {
      int e = tid + i * 128;
      int n = e >> 3, k4 = e & 7;
      float4 v = *(const float4*)(W + (long)(colbase + n) * K + kt * 32 + k4 * 4);
      st4h(&Bs[n][k4 * 4], v, 1.0f);
    }
    __syncthreads();
    v16h afrag;
    for (int e = 0; e < 16; ++e) afrag[e] = As[wave * 16 + nlane][kA_idx(hi, e)];
    for (int nt = 0; nt < 8; ++nt) {
      v16h bfrag;
      for (int e = 0; e < 16; ++e) bfrag[e] = Bs[nt * 16 + nlane][hi * 16 + e];
      acc[nt] = __builtin_amdgcn_wmma_f32_16x16x32_f16(false, afrag, false, bfrag,
                                                       (short)0, acc[nt], false, false);
    }
    __syncthreads();
  }
  for (int nt = 0; nt < 8; ++nt) {
    int colg = colbase + nt * 16 + nlane;
    float badd = 0.0f;
    if (bias0) badd += bias0[colg];
    if (bias1) badd += bias1[colg];
    for (int r = 0; r < 8; ++r) {
      int row = rowbase + wave * 16 + hi * 8 + r;
      if (row < M) {
        long idx = (long)row * ldc + colg;
        float v = acc[nt][r] + badd;
        if (accumulate) v += C[idx];
        if (do_relu) v = fmaxf(v, 0.0f);
        C[idx] = v;
      }
    }
  }
}

// ---------------- fused node head: softmax(relu(z@W1^T+b1)@W2^T+b2) -> out[M,2]
__global__ __launch_bounds__(128) void node_head_wmma(
    const float* __restrict__ z, const float* __restrict__ W1, const float* __restrict__ b1v,
    const float* __restrict__ W2, const float* __restrict__ b2v, float* __restrict__ out, int M) {
  __shared__ _Float16 As[64][40];
  __shared__ _Float16 Bs[128][40];
  __shared__ float ts[64][128];
  __shared__ float lg[64][2];
  const int tid = threadIdx.x;
  const int lane = tid & 31;
  const int wave = tid >> 5;
  const int hi = lane >> 4;
  const int nlane = lane & 15;
  const int rowbase = blockIdx.x * 64;
  const int Mc = M - 1;

  v8f acc[8];
  for (int i = 0; i < 8; ++i)
    for (int j = 0; j < 8; ++j) acc[i][j] = 0.0f;

  for (int kt = 0; kt < 4; ++kt) {
    for (int i = 0; i < 4; ++i) {
      int e = tid + i * 128;
      int r = e >> 3, k4 = e & 7;
      int row = rowbase + r;
      int rc = (row < M) ? row : Mc;
      float4 v = *(const float4*)(z + (long)rc * HDIM + kt * 32 + k4 * 4);
      st4h(&As[r][k4 * 4], v, 1.0f);
    }
    for (int i = 0; i < 8; ++i) {
      int e = tid + i * 128;
      int n = e >> 3, k4 = e & 7;
      float4 v = *(const float4*)(W1 + (long)n * HDIM + kt * 32 + k4 * 4);
      st4h(&Bs[n][k4 * 4], v, 1.0f);
    }
    __syncthreads();
    v16h afrag;
    for (int e = 0; e < 16; ++e) afrag[e] = As[wave * 16 + nlane][kA_idx(hi, e)];
    for (int nt = 0; nt < 8; ++nt) {
      v16h bfrag;
      for (int e = 0; e < 16; ++e) bfrag[e] = Bs[nt * 16 + nlane][hi * 16 + e];
      acc[nt] = __builtin_amdgcn_wmma_f32_16x16x32_f16(false, afrag, false, bfrag,
                                                       (short)0, acc[nt], false, false);
    }
    __syncthreads();
  }
  for (int nt = 0; nt < 8; ++nt) {
    int c = nt * 16 + nlane;
    float bb = b1v[c];
    for (int r = 0; r < 8; ++r) {
      int lr = wave * 16 + hi * 8 + r;
      ts[lr][c] = fmaxf(acc[nt][r] + bb, 0.0f);
    }
  }
  __syncthreads();
  {
    int lr = tid >> 1, cls = tid & 1;
    float s = b2v[cls];
    for (int c = 0; c < HDIM; ++c) s += ts[lr][c] * W2[cls * HDIM + c];
    lg[lr][cls] = s;
  }
  __syncthreads();
  {
    int lr = tid >> 1, cls = tid & 1;
    int ng = rowbase + lr;
    if (ng < M) {
      float l0 = lg[lr][0], l1 = lg[lr][1];
      float m = fmaxf(l0, l1);
      float e0 = expf(l0 - m), e1 = expf(l1 - m);
      out[(long)ng * 2 + cls] = ((cls == 0) ? e0 : e1) / (e0 + e1);
    }
  }
}

// ---------------- fused edge head: ef=[z[row],z[col],ea] (K=264, padded to 288)
__global__ __launch_bounds__(128) void edge_head_wmma(
    const float* __restrict__ z, const int* __restrict__ rowi, const int* __restrict__ coli,
    const float* __restrict__ ea, const float* __restrict__ W1, const float* __restrict__ b1v,
    const float* __restrict__ W2, const float* __restrict__ b2v, float* __restrict__ out, int E) {
  __shared__ _Float16 As[64][40];
  __shared__ _Float16 Bs[128][40];
  __shared__ float ts[64][128];
  __shared__ float lg[64][2];
  const int tid = threadIdx.x;
  const int lane = tid & 31;
  const int wave = tid >> 5;
  const int hi = lane >> 4;
  const int nlane = lane & 15;
  const int ebase = blockIdx.x * 64;
  const int Ec = E - 1;

  v8f acc[8];
  for (int i = 0; i < 8; ++i)
    for (int j = 0; j < 8; ++j) acc[i][j] = 0.0f;

  for (int kt = 0; kt < 9; ++kt) {
    // ---- stage A tile: source region is uniform per kt ----
    if (kt < 8) {
      const int* __restrict__ idxp = (kt < 4) ? rowi : coli;
      const int foff = (kt & 3) * 32;
      for (int i = 0; i < 4; ++i) {
        int e = tid + i * 128;
        int r = e >> 3, k4 = e & 7;
        int eg = ebase + r;
        int egc = (eg < E) ? eg : Ec;        // clamp; OOB rows harmless
        long node = idxp[egc];
        float4 v = *(const float4*)(z + node * HDIM + foff + k4 * 4);
        st4h(&As[r][k4 * 4], v, 1.0f);
      }
    } else {
      for (int i = 0; i < 4; ++i) {
        int e = tid + i * 128;
        int r = e >> 3, k4 = e & 7;
        int eg = ebase + r;
        int egc = (eg < E) ? eg : Ec;
        // only first 8 of 32 k are edge_attr; rest zero (B is zero there anyway)
        float4 v = *(const float4*)(ea + (long)egc * 8 + (k4 & 1) * 4);
        float m = (k4 < 2) ? 1.0f : 0.0f;
        st4h(&As[r][k4 * 4], v, m);
      }
    }
    // ---- stage B tile (zero-pad K >= 264) ----
    for (int i = 0; i < 8; ++i) {
      int e = tid + i * 128;
      int n = e >> 3, k4 = e & 7;
      int kk = kt * 32 + k4 * 4;
      int kkc = (kk < 264) ? kk : 260;       // clamped valid address
      float4 v = *(const float4*)(W1 + (long)n * 264 + kkc);
      float m = (kk < 264) ? 1.0f : 0.0f;
      st4h(&Bs[n][k4 * 4], v, m);
    }
    __syncthreads();
    v16h afrag;
    for (int e = 0; e < 16; ++e) afrag[e] = As[wave * 16 + nlane][kA_idx(hi, e)];
    for (int nt = 0; nt < 8; ++nt) {
      v16h bfrag;
      for (int e = 0; e < 16; ++e) bfrag[e] = Bs[nt * 16 + nlane][hi * 16 + e];
      acc[nt] = __builtin_amdgcn_wmma_f32_16x16x32_f16(false, afrag, false, bfrag,
                                                       (short)0, acc[nt], false, false);
    }
    __syncthreads();
  }
  for (int nt = 0; nt < 8; ++nt) {
    int c = nt * 16 + nlane;
    float bb = b1v[c];
    for (int r = 0; r < 8; ++r) {
      int lr = wave * 16 + hi * 8 + r;
      ts[lr][c] = fmaxf(acc[nt][r] + bb, 0.0f);
    }
  }
  __syncthreads();
  {
    int lr = tid >> 1, cls = tid & 1;
    float s = b2v[cls];
    for (int c = 0; c < HDIM; ++c) s += ts[lr][c] * W2[cls * HDIM + c];
    lg[lr][cls] = s;
  }
  __syncthreads();
  {
    int lr = tid >> 1, cls = tid & 1;
    int eg = ebase + lr;
    if (eg < E) {
      float l0 = lg[lr][0], l1 = lg[lr][1];
      float m = fmaxf(l0, l1);
      float e0 = expf(l0 - m), e1 = expf(l1 - m);
      out[(long)eg * 2 + cls] = ((cls == 0) ? e0 : e1) / (e0 + e1);
    }
  }
}

// ---------------- host ----------------
extern "C" void kernel_launch(void* const* d_in, const int* in_sizes, int n_in,
                              void* d_out, int out_size, void* d_ws, size_t ws_size,
                              hipStream_t stream) {
  const float* x     = (const float*)d_in[0];
  const float* ea    = (const float*)d_in[1];
  const float* w1    = (const float*)d_in[2];
  const float* we1   = (const float*)d_in[3];
  const float* as1   = (const float*)d_in[4];
  const float* ad1   = (const float*)d_in[5];
  const float* ae1   = (const float*)d_in[6];
  const float* b1    = (const float*)d_in[7];
  const float* cw[2]  = {(const float*)d_in[8],  (const float*)d_in[12]};
  const float* cas[2] = {(const float*)d_in[9],  (const float*)d_in[13]};
  const float* cad[2] = {(const float*)d_in[10], (const float*)d_in[14]};
  const float* cb[2]  = {(const float*)d_in[11], (const float*)d_in[15]};
  const float* lwih_f = (const float*)d_in[16];
  const float* lwhh_f = (const float*)d_in[17];
  const float* lbih_f = (const float*)d_in[18];
  const float* lbhh_f = (const float*)d_in[19];
  const float* lwih_b = (const float*)d_in[20];
  const float* lwhh_b = (const float*)d_in[21];
  const float* lbih_b = (const float*)d_in[22];
  const float* lbhh_b = (const float*)d_in[23];
  const float* jaw    = (const float*)d_in[24];
  const float* jab    = (const float*)d_in[25];
  const float* flw    = (const float*)d_in[26];
  const float* flb    = (const float*)d_in[27];
  const float* ncw1   = (const float*)d_in[28];
  const float* ncb1   = (const float*)d_in[29];
  const float* ncw2   = (const float*)d_in[30];
  const float* ncb2   = (const float*)d_in[31];
  const float* ecw1   = (const float*)d_in[32];
  const float* ecb1   = (const float*)d_in[33];
  const float* ecw2   = (const float*)d_in[34];
  const float* ecb2   = (const float*)d_in[35];
  const int*   eidx   = (const int*)d_in[36];

  const int N = in_sizes[0] / 16;
  const int E = in_sizes[1] / 8;
  const int* rowi = eidx;
  const int* coli = eidx + E;

  // workspace carve-up
  char* ws = (char*)d_ws;
  size_t off = 0;
  auto carve = [&](size_t bytes) -> void* {
    void* p = (void*)(ws + off);
    off += (bytes + 255) & ~(size_t)255;
    return p;
  };
  float*    h1    = (float*)carve((size_t)N * HDIM * 4);
  float*    xs0   = (float*)carve((size_t)N * HDIM * 4);
  float*    xs1   = (float*)carve((size_t)N * HDIM * 4);
  float*    zbuf  = (float*)carve((size_t)N * HDIM * 4);
  float*    hbuf  = (float*)carve((size_t)N * HDIM * 4);
  float*    cbuf  = (float*)carve((size_t)N * HDIM * 4);
  float*    gbuf  = (float*)carve((size_t)N * 512 * 4);
  float*    ebuf  = (float*)carve((size_t)E * 4);
  unsigned* mord  = (unsigned*)carve((size_t)N * 4);
  float*    denom = (float*)carve((size_t)N * 4);
  float*    sd    = (float*)carve((size_t)2 * N * 4);
  float*    score = (float*)carve((size_t)2 * N * 4);
  float*    vsm   = (float*)carve(64);

  const int TB = 256;
  const long NH = (long)N * HDIM;
  dim3 gN((unsigned)((NH + TB - 1) / TB));
  dim3 gNn((unsigned)((N + TB - 1) / TB));
  dim3 gE((unsigned)((E + TB - 1) / TB));
  dim3 gEH((unsigned)(((long)E * HDIM + TB - 1) / TB));
  dim3 gemmG((unsigned)((N + 63) / 64), 1);
  dim3 gemmG4((unsigned)((N + 63) / 64), 4);

  // ===== encode GAT (src=row -> dst=col, with edge attrs) =====
  compute_vsmall<<<1, 8, 0, stream>>>(we1, ae1, vsm);
  encode_gemm<<<gN, TB, 0, stream>>>(x, w1, hbuf, N);
  node_dots<<<gNn, TB, 0, stream>>>(hbuf, as1, ad1, sd, N);
  fill_u32<<<gNn, TB, 0, stream>>>(mord, 0u, N);
  fill_f32<<<gNn, TB, 0, stream>>>(denom, 0.0f, N);
  fill_f32<<<gN, TB, 0, stream>>>(h1, 0.0f, NH);
  edge_score<<<gE, TB, 0, stream>>>(ebuf, sd, rowi, coli, ea, vsm, mord, N, E);
  edge_exp<<<gE, TB, 0, stream>>>(ebuf, coli, mord, denom, E);
  edge_agg<<<gEH, TB, 0, stream>>>(h1, hbuf, rowi, coli, ebuf, denom, E);
  bias_act<<<gN, TB, 0, stream>>>(h1, b1, NH, 1);

  // ===== core GAT layers (src=col -> dst=row) =====
  const float* zin = h1;
  float* zouts[2] = {xs0, xs1};
  for (int l = 0; l < 2; ++l) {
    gemm128_wmma<<<gemmG, 128, 0, stream>>>(zin, cw[l], nullptr, nullptr, hbuf, N, 128, 128, 0, 0);
    node_dots<<<gNn, TB, 0, stream>>>(hbuf, cas[l], cad[l], sd, N);
    fill_u32<<<gNn, TB, 0, stream>>>(mord, 0u, N);
    fill_f32<<<gNn, TB, 0, stream>>>(denom, 0.0f, N);
    fill_f32<<<gN, TB, 0, stream>>>(zouts[l], 0.0f, NH);
    edge_score<<<gE, TB, 0, stream>>>(ebuf, sd, coli, rowi, nullptr, nullptr, mord, N, E);
    edge_exp<<<gE, TB, 0, stream>>>(ebuf, rowi, mord, denom, E);
    edge_agg<<<gEH, TB, 0, stream>>>(zouts[l], hbuf, coli, rowi, ebuf, denom, E);
    bias_act<<<gN, TB, 0, stream>>>(zouts[l], cb[l], NH, 1);
    zin = zouts[l];
  }

  // ===== JumpingKnowledge bi-LSTM (L=2) with incremental attention scores =====
  fill_from<<<gNn, TB, 0, stream>>>(score, jab, N);
  fill_from<<<gNn, TB, 0, stream>>>(score + N, jab, N);

  // forward: xs0 then xs1
  fill_f32<<<gN, TB, 0, stream>>>(cbuf, 0.0f, NH);
  gemm128_wmma<<<gemmG4, 128, 0, stream>>>(xs0, lwih_f, lbih_f, lbhh_f, gbuf, N, 128, 512, 0, 0);
  lstm_gates<<<gN, TB, 0, stream>>>(gbuf, cbuf, hbuf, N);
  score_acc<<<gNn, TB, 0, stream>>>(score, hbuf, jaw, N);
  gemm128_wmma<<<gemmG4, 128, 0, stream>>>(xs1, lwih_f, lbih_f, lbhh_f, gbuf, N, 128, 512, 0, 0);
  gemm128_wmma<<<gemmG4, 128, 0, stream>>>(hbuf, lwhh_f, nullptr, nullptr, gbuf, N, 128, 512, 1, 0);
  lstm_gates<<<gN, TB, 0, stream>>>(gbuf, cbuf, hbuf, N);
  score_acc<<<gNn, TB, 0, stream>>>(score + N, hbuf, jaw, N);

  // backward: xs1 then xs0
  fill_f32<<<gN, TB, 0, stream>>>(cbuf, 0.0f, NH);
  gemm128_wmma<<<gemmG4, 128, 0, stream>>>(xs1, lwih_b, lbih_b, lbhh_b, gbuf, N, 128, 512, 0, 0);
  lstm_gates<<<gN, TB, 0, stream>>>(gbuf, cbuf, hbuf, N);
  score_acc<<<gNn, TB, 0, stream>>>(score + N, hbuf, jaw + 128, N);
  gemm128_wmma<<<gemmG4, 128, 0, stream>>>(xs0, lwih_b, lbih_b, lbhh_b, gbuf, N, 128, 512, 0, 0);
  gemm128_wmma<<<gemmG4, 128, 0, stream>>>(hbuf, lwhh_b, nullptr, nullptr, gbuf, N, 128, 512, 1, 0);
  lstm_gates<<<gN, TB, 0, stream>>>(gbuf, cbuf, hbuf, N);
  score_acc<<<gNn, TB, 0, stream>>>(score, hbuf, jaw + 128, N);

  // attention softmax over layers + mix, then final linear
  zmix_kernel<<<gN, TB, 0, stream>>>(score, xs0, xs1, hbuf, N);
  gemm128_wmma<<<gemmG, 128, 0, stream>>>(hbuf, flw, flb, nullptr, zbuf, N, 128, 128, 0, 0);

  // ===== heads =====
  float* out = (float*)d_out;
  node_head_wmma<<<dim3((unsigned)((N + 63) / 64)), 128, 0, stream>>>(
      zbuf, ncw1, ncb1, ncw2, ncb2, out, N);
  edge_head_wmma<<<dim3((unsigned)((E + 63) / 64)), 128, 0, stream>>>(
      zbuf, rowi, coli, ea, ecw1, ecb1, ecw2, ecb2, out + (long)N * 2, E);
}
